// mamba_8993661518621
// MI455X (gfx1250) — matrix operand order repaired and checked by reading, not compiled, not run on hardware
//
#include <hip/hip_runtime.h>
#include <hip/hip_bf16.h>
#include <stdint.h>

// Problem constants (match reference)
#define B_      8
#define T_      1024
#define IN_DIM_ 32
#define HID_    512
#define L_      2
#define DI_     1024
#define DS_     16
#define KW_     4
#define DTR_    32
#define F_      96
#define M_      (B_*T_)      // 8192 rows
#define PROJ_   64           // DTR + 2*DS

#define BSTRIDE 40           // padded LDS row stride (elements) for 64x32 B tile

typedef __attribute__((ext_vector_type(16))) __bf16 v16bf;
typedef __attribute__((ext_vector_type(8)))  __bf16 v8bf;
typedef __attribute__((ext_vector_type(8)))  float  v8f;
typedef int v4i __attribute__((vector_size(16)));   // matches async-LDS builtin param

#if defined(__has_builtin)
#  if __has_builtin(__builtin_amdgcn_global_load_async_to_lds_b128)
#    define HAVE_ASYNC_LDS 1
#  else
#    define HAVE_ASYNC_LDS 0
#  endif
#else
#  define HAVE_ASYNC_LDS 0
#endif

__device__ __forceinline__ float bf2f(unsigned short u) {
    unsigned v = ((unsigned)u) << 16;
    float f; __builtin_memcpy(&f, &v, 4); return f;
}
__device__ __forceinline__ unsigned short f2bf(float f) {
    unsigned u; __builtin_memcpy(&u, &f, 4);
    u += 0x7FFFu + ((u >> 16) & 1u);           // round-to-nearest-even
    return (unsigned short)(u >> 16);
}

__device__ __forceinline__ v16bf cat16(v8bf lo, v8bf hi) {
    return __builtin_shufflevector(lo, hi, 0,1,2,3,4,5,6,7,8,9,10,11,12,13,14,15);
}

// A fragment: lane L -> row M=L&15; K {lh*16..+15}
__device__ __forceinline__ v16bf loadA(const __bf16* __restrict__ row, int k, int lh) {
    v8bf lo = *(const v8bf*)(row + k + lh * 16);
    v8bf hi = *(const v8bf*)(row + k + lh * 16 + 8);
    return cat16(lo, hi);
}

// B fragment from LDS tile (row-major, padded): lane L -> col N, K {lh*16..+15}
__device__ __forceinline__ v16bf loadB(const unsigned short* bs, int n, int lh) {
    const __bf16* p = (const __bf16*)(bs + n * BSTRIDE + lh * 16);
    v8bf lo = *(const v8bf*)p;
    v8bf hi = *(const v8bf*)(p + 8);
    return cat16(lo, hi);
}

// Cooperative stage of 64(N) x 32(K) bf16 B tile into LDS (one b128 per thread).
__device__ __forceinline__ void stageB(const __bf16* __restrict__ W, int Kdim,
                                       int nBase, int k, unsigned short* dstTile,
                                       int tid) {
    int n  = tid >> 2;          // 0..63
    int kk = (tid & 3) * 8;     // 0,8,16,24
    const __bf16* src = W + (size_t)(nBase + n) * Kdim + k + kk;
    unsigned short* dst = dstTile + n * BSTRIDE + kk;
#if HAVE_ASYNC_LDS
    __builtin_amdgcn_global_load_async_to_lds_b128(
        (__attribute__((address_space(1))) v4i*)(unsigned long long)(const void*)src,
        (__attribute__((address_space(3))) v4i*)(unsigned)(unsigned long long)(void*)dst,
        0, 0);
#else
    *(v8bf*)dst = *(const v8bf*)src;
#endif
}

__device__ __forceinline__ void stage_wait() {
#if HAVE_ASYNC_LDS
#  if __has_builtin(__builtin_amdgcn_s_wait_asynccnt)
    __builtin_amdgcn_s_wait_asynccnt(0);
#  else
    asm volatile("s_wait_asynccnt 0x0" ::: "memory");
#  endif
#endif
}

// ---------------- elementwise conversion kernels ----------------
__global__ void cvt_f32_bf16(const float* __restrict__ in,
                             unsigned short* __restrict__ out, long n) {
    long i = (long)blockIdx.x * blockDim.x + threadIdx.x;
    if (i < n) out[i] = f2bf(in[i]);
}

__global__ void cvt_x_scaled(const float* __restrict__ x,
                             unsigned short* __restrict__ out) {
    int i = blockIdx.x * blockDim.x + threadIdx.x;   // M_ * IN_DIM_
    if (i >= M_ * IN_DIM_) return;
    int c = i & (IN_DIM_ - 1);
    float v = x[i];
    if (c == 0) v *= 0.3f; else if (c == 1) v *= 0.5f; else if (c == 2) v *= 0.9f;
    out[i] = f2bf(v);
}

// ---------------- WMMA GEMM: C(M,N) = A(M,K) @ W(N,K)^T ----------------
// Block: 256 thr = 8 waves; wave computes 32(M)x64(N); block covers 256(M)x64(N).
// B tile (shared by all 8 waves) is double-buffered in LDS via async-to-LDS.
// mode 0: store (opt. +bias); mode 1: softplus(acc+bias) -> f32; mode 2: outF += acc
__global__ void __launch_bounds__(256)
wmma_gemm(const __bf16* __restrict__ A, const __bf16* __restrict__ W,
          const float* __restrict__ bias,
          float* __restrict__ outF, unsigned short* __restrict__ outB,
          int Mdim, int Ndim, int Kdim, int mode)
{
    __shared__ unsigned short Bs[2][64 * BSTRIDE];

    const int tid   = threadIdx.x;
    const int wave  = tid >> 5;
    const int lane  = tid & 31;
    const int lh    = lane >> 4;
    const int ll    = lane & 15;
    const int mBase = blockIdx.x * 256 + wave * 32;
    const int nBase = blockIdx.y * 64;

    v8f acc[2][4];
    #pragma unroll
    for (int mi = 0; mi < 2; ++mi)
        #pragma unroll
        for (int t = 0; t < 4; ++t) acc[mi][t] = (v8f){};

    const __bf16* aRow0 = A + (size_t)(mBase + ll) * Kdim;
    const __bf16* aRow1 = A + (size_t)(mBase + 16 + ll) * Kdim;

    const int nk = Kdim >> 5;           // K-steps of 32
    stageB(W, Kdim, nBase, 0, Bs[0], tid);
    v16bf a0 = loadA(aRow0, 0, lh);
    v16bf a1 = loadA(aRow1, 0, lh);

    for (int kt = 0; kt < nk; ++kt) {
        stage_wait();
        __syncthreads();                 // B tile ready; prev buffer free

        const bool more = (kt + 1 < nk);
        if (more) stageB(W, Kdim, nBase, (kt + 1) * 32, Bs[(kt + 1) & 1], tid);

        v16bf na0 = a0, na1 = a1;
        if (more) {
            int kn = (kt + 1) * 32;
            if (kn + 32 < Kdim) __builtin_prefetch(aRow0 + kn + 32, 0, 3);
            na0 = loadA(aRow0, kn, lh);
            na1 = loadA(aRow1, kn, lh);
        }

        const unsigned short* bs = Bs[kt & 1];
        v16bf b0 = loadB(bs,  0 + ll, lh);
        v16bf b1 = loadB(bs, 16 + ll, lh);
        v16bf b2 = loadB(bs, 32 + ll, lh);
        v16bf b3 = loadB(bs, 48 + ll, lh);

        acc[0][0] = __builtin_amdgcn_wmma_f32_16x16x32_bf16(false, a0, false, b0, (short)0, acc[0][0], false, false);
        acc[0][1] = __builtin_amdgcn_wmma_f32_16x16x32_bf16(false, a0, false, b1, (short)0, acc[0][1], false, false);
        acc[0][2] = __builtin_amdgcn_wmma_f32_16x16x32_bf16(false, a0, false, b2, (short)0, acc[0][2], false, false);
        acc[0][3] = __builtin_amdgcn_wmma_f32_16x16x32_bf16(false, a0, false, b3, (short)0, acc[0][3], false, false);
        acc[1][0] = __builtin_amdgcn_wmma_f32_16x16x32_bf16(false, a1, false, b0, (short)0, acc[1][0], false, false);
        acc[1][1] = __builtin_amdgcn_wmma_f32_16x16x32_bf16(false, a1, false, b1, (short)0, acc[1][1], false, false);
        acc[1][2] = __builtin_amdgcn_wmma_f32_16x16x32_bf16(false, a1, false, b2, (short)0, acc[1][2], false, false);
        acc[1][3] = __builtin_amdgcn_wmma_f32_16x16x32_bf16(false, a1, false, b3, (short)0, acc[1][3], false, false);

        a0 = na0; a1 = na1;
    }

    // D layout: vgpr r, lane L -> (M = r + 8*lh, N = ll)
    #pragma unroll
    for (int mi = 0; mi < 2; ++mi) {
        #pragma unroll
        for (int r = 0; r < 8; ++r) {
            int m = mBase + mi * 16 + r + lh * 8;
            #pragma unroll
            for (int t = 0; t < 4; ++t) {
                int n = nBase + t * 16 + ll;
                float v = acc[mi][t][r];
                if (bias) v += bias[n];
                if (mode == 1) v = (v > 20.f) ? v : log1pf(__expf(v));
                size_t o = (size_t)m * Ndim + n;
                if (outF) { if (mode == 2) outF[o] += v; else outF[o] = v; }
                if (outB) outB[o] = f2bf(v);
            }
        }
    }
}

// ---------------- RMSNorm -> bf16 ----------------
__global__ void __launch_bounds__(128)
rmsnorm_bf16(const float* __restrict__ h, const float* __restrict__ w,
             unsigned short* __restrict__ out)
{
    int row = blockIdx.x, tid = threadIdx.x;
    const float* hr = h + (size_t)row * HID_;
    float v[4]; float ss = 0.f;
    #pragma unroll
    for (int i = 0; i < 4; ++i) { v[i] = hr[tid + i * 128]; ss += v[i] * v[i]; }
    #pragma unroll
    for (int off = 16; off > 0; off >>= 1) ss += __shfl_xor(ss, off, 32);
    __shared__ float red[4];
    if ((tid & 31) == 0) red[tid >> 5] = ss;
    __syncthreads();
    float inv = rsqrtf((red[0] + red[1] + red[2] + red[3]) * (1.f / HID_) + 1e-5f);
    #pragma unroll
    for (int i = 0; i < 4; ++i) {
        int c = tid + i * 128;
        out[(size_t)row * HID_ + c] = f2bf(v[i] * inv * w[c]);
    }
}

// ---------------- depthwise causal conv (K=4) + SiLU ----------------
__global__ void conv_silu(const unsigned short* __restrict__ xz,
                          const float* __restrict__ cw, const float* __restrict__ cb,
                          unsigned short* __restrict__ xssm)
{
    int i = blockIdx.x * blockDim.x + threadIdx.x;       // (b*T+t)*DI + d
    if (i >= B_ * T_ * DI_) return;
    int d = i & (DI_ - 1);
    int t = (i >> 10) & (T_ - 1);
    int b = i >> 20;
    float acc = cb[d];
    #pragma unroll
    for (int j = 0; j < KW_; ++j) {
        int tt = t - (KW_ - 1) + j;
        if (tt >= 0)
            acc += cw[d * KW_ + j] * bf2f(xz[((size_t)(b * T_ + tt)) * (2 * DI_) + d]);
    }
    xssm[i] = f2bf(acc / (1.f + __expf(-acc)));          // silu
}

// ---------------- strided dt_raw extraction -> bf16 ----------------
__global__ void extract_dt(const float* __restrict__ dbl,
                           unsigned short* __restrict__ dtr)
{
    int i = blockIdx.x * blockDim.x + threadIdx.x;       // M_*DTR_
    if (i >= M_ * DTR_) return;
    int m = i >> 5, r = i & 31;
    dtr[i] = f2bf(dbl[(size_t)m * PROJ_ + r]);
}

// ---------------- selective scan: state in registers, fused D + silu(z) ----
__global__ void __launch_bounds__(256)
ssm_scan(const float* __restrict__ delta, const unsigned short* __restrict__ xssm,
         const unsigned short* __restrict__ xz, const float* __restrict__ dbl,
         const float* __restrict__ A_log, const float* __restrict__ Dp,
         unsigned short* __restrict__ y)
{
    int b = blockIdx.x;
    int d = blockIdx.y * 256 + threadIdx.x;
    float A[DS_], st[DS_];
    #pragma unroll
    for (int s = 0; s < DS_; ++s) { A[s] = -__expf(A_log[d * DS_ + s]); st[s] = 0.f; }
    float Dd = Dp[d];
    for (int t = 0; t < T_; ++t) {
        size_t m  = (size_t)b * T_ + t;
        float  dt = delta[m * DI_ + d];
        float  xv = bf2f(xssm[m * DI_ + d]);
        const float* bc = dbl + m * PROJ_;   // uniform per (b,t): broadcast loads
        float sum = 0.f;
        #pragma unroll
        for (int s = 0; s < DS_; ++s) {
            float Bv = bc[DTR_ + s];
            float Cv = bc[DTR_ + DS_ + s];
            st[s] = st[s] * __expf(dt * A[s]) + dt * Bv * xv;
            sum  += st[s] * Cv;
        }
        float zv = bf2f(xz[m * (2 * DI_) + DI_ + d]);
        float gate = zv / (1.f + __expf(-zv));
        y[m * DI_ + d] = f2bf((sum + xv * Dd) * gate);
    }
}

// ---------------- final FC on last timestep ----------------
__global__ void fc_kernel(const float* __restrict__ h, const float* __restrict__ fw,
                          const float* __restrict__ fb, float* __restrict__ out)
{
    int i = blockIdx.x * blockDim.x + threadIdx.x;       // B_*F_ = 768
    if (i >= B_ * F_) return;
    int b = i / F_, f = i % F_;
    const float* hr = h + ((size_t)(b * T_ + T_ - 1)) * HID_;
    const float* wr = fw + (size_t)f * HID_;
    float s = 0.f;
    for (int k = 0; k < HID_; ++k) s += hr[k] * wr[k];
    out[i] = s + fb[f];
}

// ---------------- host orchestration ----------------
extern "C" void kernel_launch(void* const* d_in, const int* in_sizes, int n_in,
                              void* d_out, int out_size, void* d_ws, size_t ws_size,
                              hipStream_t stream)
{
    (void)in_sizes; (void)n_in; (void)out_size; (void)ws_size;
    const float* x        = (const float*)d_in[0];
    const float* ip_w     = (const float*)d_in[1];
    const float* ip_b     = (const float*)d_in[2];
    const float* norm_w   = (const float*)d_in[3];
    const float* inp_w    = (const float*)d_in[4];
    const float* conv_w   = (const float*)d_in[5];
    const float* conv_b   = (const float*)d_in[6];
    const float* xproj_w  = (const float*)d_in[7];
    const float* dtproj_w = (const float*)d_in[8];
    const float* dtproj_b = (const float*)d_in[9];
    const float* A_log    = (const float*)d_in[10];
    const float* Dp       = (const float*)d_in[11];
    const float* outp_w   = (const float*)d_in[12];
    const float* fc_w     = (const float*)d_in[13];
    const float* fc_b     = (const float*)d_in[14];
    float* out = (float*)d_out;

    char* ws = (char*)d_ws;
    size_t off = 0;
    auto take = [&](size_t bytes) -> char* {
        char* p = ws + off;
        off = (off + bytes + 255) & ~(size_t)255;
        return p;
    };
    unsigned short* wb_ip     = (unsigned short*)take((size_t)HID_ * IN_DIM_ * 2);
    unsigned short* wb_inp    = (unsigned short*)take((size_t)L_ * 2 * DI_ * HID_ * 2);
    unsigned short* wb_xproj  = (unsigned short*)take((size_t)L_ * PROJ_ * DI_ * 2);
    unsigned short* wb_dtproj = (unsigned short*)take((size_t)L_ * DI_ * DTR_ * 2);
    unsigned short* wb_outp   = (unsigned short*)take((size_t)L_ * HID_ * DI_ * 2);
    unsigned short* xb        = (unsigned short*)take((size_t)M_ * IN_DIM_ * 2);
    float*          hbuf      = (float*)take((size_t)M_ * HID_ * 4);
    unsigned short* hn        = (unsigned short*)take((size_t)M_ * HID_ * 2);
    unsigned short* xzb       = (unsigned short*)take((size_t)M_ * 2 * DI_ * 2);
    unsigned short* xssmb     = (unsigned short*)take((size_t)M_ * DI_ * 2);
    float*          dblf      = (float*)take((size_t)M_ * PROJ_ * 4);
    unsigned short* dtrb      = (unsigned short*)take((size_t)M_ * DTR_ * 2);
    float*          deltaf    = (float*)take((size_t)M_ * DI_ * 4);
    unsigned short* yb        = (unsigned short*)take((size_t)M_ * DI_ * 2);

    auto cvt = [&](const float* in, unsigned short* o, long n) {
        cvt_f32_bf16<<<dim3((unsigned)((n + 255) / 256)), dim3(256), 0, stream>>>(in, o, n);
    };
    cvt(ip_w,     wb_ip,     (long)HID_ * IN_DIM_);
    cvt(inp_w,    wb_inp,    (long)L_ * 2 * DI_ * HID_);
    cvt(xproj_w,  wb_xproj,  (long)L_ * PROJ_ * DI_);
    cvt(dtproj_w, wb_dtproj, (long)L_ * DI_ * DTR_);
    cvt(outp_w,   wb_outp,   (long)L_ * HID_ * DI_);
    cvt_x_scaled<<<dim3((M_ * IN_DIM_ + 255) / 256), dim3(256), 0, stream>>>(x, xb);

    // h = x @ ip_w^T + ip_b
    wmma_gemm<<<dim3(M_ / 256, HID_ / 64), dim3(256), 0, stream>>>(
        (const __bf16*)xb, (const __bf16*)wb_ip, ip_b, hbuf, nullptr,
        M_, HID_, IN_DIM_, 0);

    for (int l = 0; l < L_; ++l) {
        rmsnorm_bf16<<<dim3(M_), dim3(128), 0, stream>>>(
            hbuf, norm_w + (size_t)l * HID_, hn);
        // xz = hn @ inp_w^T  (bf16 out)
        wmma_gemm<<<dim3(M_ / 256, (2 * DI_) / 64), dim3(256), 0, stream>>>(
            (const __bf16*)hn, (const __bf16*)(wb_inp + (size_t)l * 2 * DI_ * HID_),
            nullptr, nullptr, xzb, M_, 2 * DI_, HID_, 0);
        conv_silu<<<dim3((B_ * T_ * DI_) / 256), dim3(256), 0, stream>>>(
            xzb, conv_w + (size_t)l * DI_ * KW_, conv_b + (size_t)l * DI_, xssmb);
        // dbl = x_ssm @ xproj_w^T  (f32 out)
        wmma_gemm<<<dim3(M_ / 256, PROJ_ / 64), dim3(256), 0, stream>>>(
            (const __bf16*)xssmb, (const __bf16*)(wb_xproj + (size_t)l * PROJ_ * DI_),
            nullptr, dblf, nullptr, M_, PROJ_, DI_, 0);
        extract_dt<<<dim3((M_ * DTR_) / 256), dim3(256), 0, stream>>>(dblf, dtrb);
        // delta = softplus(dt_raw @ dtproj_w^T + dtproj_b)
        wmma_gemm<<<dim3(M_ / 256, DI_ / 64), dim3(256), 0, stream>>>(
            (const __bf16*)dtrb, (const __bf16*)(wb_dtproj + (size_t)l * DI_ * DTR_),
            dtproj_b + (size_t)l * DI_, deltaf, nullptr, M_, DI_, DTR_, 1);
        ssm_scan<<<dim3(B_, DI_ / 256), dim3(256), 0, stream>>>(
            deltaf, xssmb, xzb, dblf,
            A_log + (size_t)l * DI_ * DS_, Dp + (size_t)l * DI_, yb);
        // h += y @ outp_w^T
        wmma_gemm<<<dim3(M_ / 256, HID_ / 64), dim3(256), 0, stream>>>(
            (const __bf16*)yb, (const __bf16*)(wb_outp + (size_t)l * HID_ * DI_),
            nullptr, hbuf, nullptr, M_, HID_, DI_, 2);
    }
    fc_kernel<<<dim3(3), dim3(256), 0, stream>>>(hbuf, fc_w, fc_b, out);
}
